// HRRPhasorBlock_84825604096564
// MI455X (gfx1250) — matrix operand order repaired and checked by reading, not compiled, not run on hardware
//
#include <hip/hip_runtime.h>
#include <hip/hip_bf16.h>

// ---------------------------------------------------------------------------
// HRR phasor block for MI455X (gfx1250, wave32, WMMA + TDM).
//   1. split f32 -> bf16(hi)+bf16(lo) for x, Wk, Wv, Wo
//   2. K = x@Wk^T+bk, V = x@Wv^T+bv via v_wmma_f32_16x16x32_bf16, tiles staged
//      into LDS by TENSOR_LOAD_TO_LDS (double buffered, s_wait_tensorcnt)
//   3. per token: L2-normalize K row, Z=K+iV, 1024-pt LDS FFT, hermitian split
//      -> Kf, Bf = Kf*Vf  (bind in frequency domain)
//   4. cumsum over L in frequency domain (chunk sums -> prefix -> apply),
//      fused with unbind:  Rf = cumsum(Bf) * conj(Kf)
//   5. per token: hermitian inverse FFT, /sqrt(pos), LayerNorm, bf16 split
//   6. out = x + r_ln @ Wo^T + bo  via WMMA
// ---------------------------------------------------------------------------

typedef __attribute__((ext_vector_type(16))) __bf16   v16bf;
typedef __attribute__((ext_vector_type(8)))  float    v8f;
typedef __attribute__((ext_vector_type(4)))  unsigned uv4;
typedef __attribute__((ext_vector_type(8)))  int      iv8;
typedef __attribute__((ext_vector_type(4)))  int      iv4;

#define DIMN   1024
#define NBINS  513
#define NCHNK  16

union FragBF { v16bf v; unsigned u[8]; };

__device__ __forceinline__ v8f wmma_bf16(v16bf a, v16bf b, v8f c) {
  return __builtin_amdgcn_wmma_f32_16x16x32_bf16(false, a, false, b,
                                                 (short)0, c, false, false);
}

#if __has_builtin(__builtin_amdgcn_tensor_load_to_lds) && \
    __has_builtin(__builtin_amdgcn_s_wait_tensorcnt)
#define USE_TDM 1
#else
#define USE_TDM 0
#endif

#if USE_TDM
// ---------------------------------------------------------------------------
// Issue one TDM descriptor: 64 rows x 32 bf16 tile from a row-major tensor
// with 1024-element row stride, DMA'd to LDS at byte offset ldsOff.
// D# layout per CDNA5 ISA ch.8 (CDNA5 ISA).
// ---------------------------------------------------------------------------
__device__ __forceinline__ void tdm_load_tile_64x32(unsigned ldsOff,
                                                    const void* gaddr) {
  unsigned long long ga = (unsigned long long)gaddr;
  uv4 g0;
  g0[0] = 1u;                                   // count=1, user mode
  g0[1] = ldsOff;                               // lds_addr (bytes)
  g0[2] = (unsigned)(ga & 0xffffffffu);         // global_addr[31:0]
  g0[3] = (unsigned)((ga >> 32) & 0x01ffffffu)  // global_addr[56:32]
          | 0x80000000u;                        // type=2 (bits 127:126)
  iv8 g1;
  g1[0] = (int)(1u << 16);          // data_size=1 (2 bytes/elem)
  g1[1] = (int)(1024u << 16);       // tensor_dim0[15:0]  @ bits 63:48
  g1[2] = (int)(1024u << 16);       // tensor_dim1[15:0]  @ bits 95:80
  g1[3] = (int)(32u << 16);         // tile_dim0 = 32     @ bits 127:112
  g1[4] = 64;                       // tile_dim1 = 64     @ bits 143:128
  g1[5] = 1024;                     // tensor_dim0_stride @ bits 191:160
  g1[6] = 0;
  g1[7] = 0;
  iv4 gz = {0, 0, 0, 0};
#if __clang_major__ >= 23
  iv8 gz8 = {0, 0, 0, 0, 0, 0, 0, 0};
  __builtin_amdgcn_tensor_load_to_lds(g0, g1, gz, gz, gz8, 0);
#else
  __builtin_amdgcn_tensor_load_to_lds(g0, g1, gz, gz, 0);
#endif
}
#endif  // USE_TDM

// ---------------------------------------------------------------------------
// f32 -> bf16 hi/lo split
// ---------------------------------------------------------------------------
__global__ void k_split_bf16(const float* __restrict__ src,
                             __bf16* __restrict__ hi,
                             __bf16* __restrict__ lo, int n) {
  int i = blockIdx.x * blockDim.x + threadIdx.x;
  if (i < n) {
    float v = src[i];
    __bf16 h = (__bf16)v;
    hi[i] = h;
    lo[i] = (__bf16)(v - (float)h);
  }
}

// ---------------------------------------------------------------------------
// Split-bf16 WMMA GEMM: out[n,d] = sum_k A[n,k]*W[d,k] + bias[d] (+ addsrc)
// 64x64 output tile / 256-thread WG (8 waves); TDM double-buffered tiles.
// ---------------------------------------------------------------------------
__global__ __launch_bounds__(256) void k_gemm_split(
    const __bf16* __restrict__ Ahi, const __bf16* __restrict__ Alo,
    const __bf16* __restrict__ Whi, const __bf16* __restrict__ Wlo,
    const float* __restrict__ bias, const float* __restrict__ addsrc,
    float* __restrict__ out) {
  __shared__ unsigned lAhi[2][1024], lAlo[2][1024];
  __shared__ unsigned lBhi[2][1024], lBlo[2][1024];

  const int tid    = threadIdx.x;
  const int lane   = tid & 31;
  const int wave   = tid >> 5;
  const int l15    = lane & 15;
  const int laneHi = lane >> 4;
  const int mblk   = blockIdx.x * 64;
  const int nblk   = blockIdx.y * 64;
  const int mw     = (wave & 3) * 16;
  const int nw     = (wave >> 2) * 32;

#if USE_TDM
  // wave 0 drives the Tensor Data Mover for the whole workgroup
  auto issue_batch = [&](int buf, int k0) {
    const char* aBase = (const char*)Ahi + ((size_t)mblk * DIMN + k0) * 2;
    const char* bBase = (const char*)Whi + ((size_t)nblk * DIMN + k0) * 2;
    const char* aLo   = (const char*)Alo + ((size_t)mblk * DIMN + k0) * 2;
    const char* bLo   = (const char*)Wlo + ((size_t)nblk * DIMN + k0) * 2;
    tdm_load_tile_64x32((unsigned)(unsigned long long)&lAhi[buf][0], aBase);
    tdm_load_tile_64x32((unsigned)(unsigned long long)&lAlo[buf][0], aLo);
    tdm_load_tile_64x32((unsigned)(unsigned long long)&lBhi[buf][0], bBase);
    tdm_load_tile_64x32((unsigned)(unsigned long long)&lBlo[buf][0], bLo);
  };
  if (wave == 0) issue_batch(0, 0);
#endif

  v8f acc0 = {};
  v8f acc1 = {};

  for (int kc = 0; kc < DIMN / 32; ++kc) {
    const int cur = kc & 1;
#if USE_TDM
    if (wave == 0) {
      if (kc + 1 < DIMN / 32) {
        issue_batch(cur ^ 1, (kc + 1) * 32);
        __builtin_amdgcn_s_wait_tensorcnt(4);  // previous batch landed
      } else {
        __builtin_amdgcn_s_wait_tensorcnt(0);
      }
    }
#else
    // fallback: manual cooperative staging
    {
      const int kd = kc * 16;  // dword offset within 512-dword row
      const unsigned* gAhi = (const unsigned*)Ahi + (size_t)mblk * 512;
      const unsigned* gAlo = (const unsigned*)Alo + (size_t)mblk * 512;
      const unsigned* gBhi = (const unsigned*)Whi + (size_t)nblk * 512;
      const unsigned* gBlo = (const unsigned*)Wlo + (size_t)nblk * 512;
      for (int idx = tid; idx < 1024; idx += 256) {
        int r = idx >> 4, c = idx & 15;
        lAhi[cur][idx] = gAhi[(size_t)r * 512 + kd + c];
        lAlo[cur][idx] = gAlo[(size_t)r * 512 + kd + c];
        lBhi[cur][idx] = gBhi[(size_t)r * 512 + kd + c];
        lBlo[cur][idx] = gBlo[(size_t)r * 512 + kd + c];
      }
    }
#endif
    __syncthreads();

    // A fragment (16x32 bf16): lane<16 K{0..7,16..23}, lane>=16 K{8..15,24..31}
    FragBF ahi, alo;
    const int arow = mw + l15;
#pragma unroll
    for (int j = 0; j < 8; ++j) {
      int k = ((j < 4) ? (2 * j) : (2 * j + 8)) + laneHi * 8;
      int d = arow * 16 + (k >> 1);
      ahi.u[j] = lAhi[cur][d];
      alo.u[j] = lAlo[cur][d];
    }
    // B fragments (32x16 bf16): lane<16 K{0..7,16..23}, lane>=16 K{8..15,24..31}
    FragBF bh0, bl0, bh1, bl1;
#pragma unroll
    for (int j = 0; j < 8; ++j) {
      int k  = ((j < 4) ? (2 * j) : (2 * j + 8)) + laneHi * 8;
      int d0 = (nw + l15) * 16 + (k >> 1);
      int d1 = (nw + 16 + l15) * 16 + (k >> 1);
      bh0.u[j] = lBhi[cur][d0];
      bl0.u[j] = lBlo[cur][d0];
      bh1.u[j] = lBhi[cur][d1];
      bl1.u[j] = lBlo[cur][d1];
    }
    acc0 = wmma_bf16(ahi.v, bh0.v, acc0);
    acc0 = wmma_bf16(ahi.v, bl0.v, acc0);
    acc0 = wmma_bf16(alo.v, bh0.v, acc0);
    acc1 = wmma_bf16(ahi.v, bh1.v, acc1);
    acc1 = wmma_bf16(ahi.v, bl1.v, acc1);
    acc1 = wmma_bf16(alo.v, bh1.v, acc1);
    __syncthreads();
  }

#pragma unroll
  for (int r = 0; r < 8; ++r) {
    int row  = mblk + mw + laneHi * 8 + r;
    int col0 = nblk + nw + l15;
    int col1 = col0 + 16;
    float v0 = acc0[r] + bias[col0];
    float v1 = acc1[r] + bias[col1];
    if (addsrc) {
      v0 += addsrc[(size_t)row * DIMN + col0];
      v1 += addsrc[(size_t)row * DIMN + col1];
    }
    out[(size_t)row * DIMN + col0] = v0;
    out[(size_t)row * DIMN + col1] = v1;
  }
}

// ---------------------------------------------------------------------------
// In-LDS 1024-point radix-2 complex FFT (input bit-reversed). dir=-1 fwd.
// ---------------------------------------------------------------------------
__device__ __forceinline__ void fft1024(float* re, float* im, int tid,
                                        float dir) {
  for (int s = 1; s <= 10; ++s) {
    const int half = 1 << (s - 1);
    for (int t2 = tid; t2 < 512; t2 += 256) {
      int j    = t2 & (half - 1);
      int idx1 = ((t2 >> (s - 1)) << s) + j;
      int idx2 = idx1 + half;
      float ang = dir * 6.283185307179586f * (float)j / (float)(1 << s);
      float sn, cs;
      sincosf(ang, &sn, &cs);
      float xr = re[idx2], xi = im[idx2];
      float tr = cs * xr - sn * xi;
      float ti = cs * xi + sn * xr;
      float ur = re[idx1], ui = im[idx1];
      re[idx2] = ur - tr; im[idx2] = ui - ti;
      re[idx1] = ur + tr; im[idx1] = ui + ti;
    }
    __syncthreads();
  }
}

// ---------------------------------------------------------------------------
// Per token: normalize K row, Z = Kn + i*V, FFT, hermitian split, bind.
// ---------------------------------------------------------------------------
__global__ __launch_bounds__(256) void k_fft_bind(
    const float* __restrict__ Kmat, const float* __restrict__ Vmat,
    float2* __restrict__ Kf, float2* __restrict__ Bf) {
  __shared__ float re[1024], im[1024], red[256];
  const int t   = blockIdx.x;
  const int tid = threadIdx.x;
  const float* krow = Kmat + (size_t)t * DIMN;
  const float* vrow = Vmat + (size_t)t * DIMN;

  float kv[4], vv[4], ss = 0.0f;
#pragma unroll
  for (int j = 0; j < 4; ++j) {
    int i = tid + j * 256;
    kv[j] = krow[i];
    vv[j] = vrow[i];
    ss += kv[j] * kv[j];
  }
  red[tid] = ss;
  __syncthreads();
  for (int s = 128; s > 0; s >>= 1) {
    if (tid < s) red[tid] += red[tid + s];
    __syncthreads();
  }
  const float scale = 1.0f / fmaxf(sqrtf(red[0]), 1e-12f);
  __syncthreads();

#pragma unroll
  for (int j = 0; j < 4; ++j) {
    int i  = tid + j * 256;
    int ri = __brev((unsigned)i) >> 22;
    re[ri] = kv[j] * scale;
    im[ri] = vv[j];
  }
  __syncthreads();
  fft1024(re, im, tid, -1.0f);

  for (int n = tid; n <= 512; n += 256) {
    int neg  = (1024 - n) & 1023;
    float zr = re[n], zi = im[n];
    float cr = re[neg], ci = -im[neg];
    float kr = 0.5f * (zr + cr), ki = 0.5f * (zi + ci);
    float dr = 0.5f * (zr - cr), di = 0.5f * (zi - ci);
    float vr = di, vi = -dr;
    size_t o = (size_t)t * NBINS + n;
    Kf[o] = make_float2(kr, ki);
    Bf[o] = make_float2(kr * vr - ki * vi, kr * vi + ki * vr);
  }
}

// ---------------------------------------------------------------------------
// Frequency-domain cumsum over L (3 passes) fused with unbind in pass 3.
// ---------------------------------------------------------------------------
__global__ void k_scan_chunk_sum(const float2* __restrict__ Bf,
                                 float2* __restrict__ csum, int L) {
  int bin = blockIdx.x * 256 + threadIdx.x;
  if (bin >= NBINS) return;
  int ch = blockIdx.y, b = blockIdx.z;
  int clen = L / NCHNK;
  size_t base = ((size_t)b * L + (size_t)ch * clen) * NBINS + bin;
  float sx = 0.0f, sy = 0.0f;
  for (int i = 0; i < clen; ++i) {
    if (i + 8 < clen)
      __builtin_prefetch(&Bf[base + (size_t)(i + 8) * NBINS], 0, 1);
    float2 v = Bf[base + (size_t)i * NBINS];
    sx += v.x; sy += v.y;
  }
  csum[((size_t)b * NCHNK + ch) * NBINS + bin] = make_float2(sx, sy);
}

__global__ void k_scan_chunk_prefix(float2* csum) {
  int bin = blockIdx.x * 256 + threadIdx.x;
  if (bin >= NBINS) return;
  int b = blockIdx.z;
  float rx = 0.0f, ry = 0.0f;
  for (int ch = 0; ch < NCHNK; ++ch) {
    size_t o = ((size_t)b * NCHNK + ch) * NBINS + bin;
    float2 v = csum[o];
    csum[o] = make_float2(rx, ry);  // exclusive prefix
    rx += v.x; ry += v.y;
  }
}

__global__ void k_scan_apply_unbind(float2* __restrict__ BfRf,
                                    const float2* __restrict__ Kf,
                                    const float2* __restrict__ csum, int L) {
  int bin = blockIdx.x * 256 + threadIdx.x;
  if (bin >= NBINS) return;
  int ch = blockIdx.y, b = blockIdx.z;
  int clen = L / NCHNK;
  float2 run = csum[((size_t)b * NCHNK + ch) * NBINS + bin];
  size_t base = ((size_t)b * L + (size_t)ch * clen) * NBINS + bin;
  for (int i = 0; i < clen; ++i) {
    size_t o = base + (size_t)i * NBINS;
    if (i + 8 < clen) {
      __builtin_prefetch(&BfRf[base + (size_t)(i + 8) * NBINS], 0, 1);
      __builtin_prefetch(&Kf[base + (size_t)(i + 8) * NBINS], 0, 1);
    }
    float2 v = BfRf[o];
    run.x += v.x; run.y += v.y;          // Mf = cumsum(Bf)
    float2 k = Kf[o];                    // Rf = Mf * conj(Kf)
    BfRf[o] = make_float2(run.x * k.x + run.y * k.y,
                          run.y * k.x - run.x * k.y);
  }
}

// ---------------------------------------------------------------------------
// Per token: hermitian inverse FFT, position norm, LayerNorm, bf16 split.
// ---------------------------------------------------------------------------
__global__ __launch_bounds__(256) void k_ifft_ln(
    const float2* __restrict__ Rf, const float* __restrict__ lng,
    const float* __restrict__ lnb, __bf16* __restrict__ rhi,
    __bf16* __restrict__ rlo, int L) {
  __shared__ float re[1024], im[1024], red[256];
  const int t   = blockIdx.x;
  const int tid = threadIdx.x;
  const int l   = t & (L - 1);

  for (int n = tid; n <= 512; n += 256) {
    float2 v = Rf[(size_t)t * NBINS + n];
    int p = __brev((unsigned)n) >> 22;
    re[p] = v.x; im[p] = v.y;
    if (n != 0 && n != 512) {
      int p2 = __brev((unsigned)(1024 - n)) >> 22;
      re[p2] = v.x; im[p2] = -v.y;
    }
  }
  __syncthreads();
  fft1024(re, im, tid, +1.0f);

  const float posn = rsqrtf((float)(l + 1)) * (1.0f / 1024.0f);
  float r[4], s1 = 0.0f, s2 = 0.0f;
#pragma unroll
  for (int j = 0; j < 4; ++j) {
    int i = tid + j * 256;
    r[j] = re[i] * posn;
    s1 += r[j];
    s2 += r[j] * r[j];
  }
  red[tid] = s1;
  __syncthreads();
  for (int s = 128; s > 0; s >>= 1) {
    if (tid < s) red[tid] += red[tid + s];
    __syncthreads();
  }
  const float mu = red[0] * (1.0f / 1024.0f);
  __syncthreads();
  red[tid] = s2;
  __syncthreads();
  for (int s = 128; s > 0; s >>= 1) {
    if (tid < s) red[tid] += red[tid + s];
    __syncthreads();
  }
  const float var  = red[0] * (1.0f / 1024.0f) - mu * mu;
  const float istd = rsqrtf(var + 1e-5f);
#pragma unroll
  for (int j = 0; j < 4; ++j) {
    int i = tid + j * 256;
    float y = (r[j] - mu) * istd * lng[i] + lnb[i];
    __bf16 h = (__bf16)y;
    rhi[(size_t)t * DIMN + i] = h;
    rlo[(size_t)t * DIMN + i] = (__bf16)(y - (float)h);
  }
}

// ---------------------------------------------------------------------------
// Host launcher
// ---------------------------------------------------------------------------
extern "C" void kernel_launch(void* const* d_in, const int* in_sizes, int n_in,
                              void* d_out, int out_size, void* d_ws,
                              size_t ws_size, hipStream_t stream) {
  (void)in_sizes; (void)n_in; (void)out_size; (void)ws_size;
  const int Bz = 4, L = 4096, N = Bz * L;

  const float* x   = (const float*)d_in[0];
  const float* Wk  = (const float*)d_in[1];
  const float* bk  = (const float*)d_in[2];
  const float* Wv  = (const float*)d_in[3];
  const float* bv  = (const float*)d_in[4];
  const float* lng = (const float*)d_in[5];
  const float* lnb = (const float*)d_in[6];
  const float* Wo  = (const float*)d_in[7];
  const float* bo  = (const float*)d_in[8];

  char* ws = (char*)d_ws;
  size_t off = 0;
  auto take = [&](size_t bytes) {
    size_t o = off;
    off += (bytes + 255) & ~(size_t)255;
    return o;
  };

  const size_t szXsplit = (size_t)N * DIMN * sizeof(__bf16);
  const size_t szSpec   = (size_t)N * NBINS * sizeof(float2);
  const size_t szMat    = (size_t)N * DIMN * sizeof(float);

  // Region A: x hi/lo splits (phase 1) then Kf (phase 2) — never co-live
  size_t regA = take(szSpec > 2 * szXsplit ? szSpec : 2 * szXsplit);
  __bf16* xhi = (__bf16*)(ws + regA);
  __bf16* xlo = xhi + (size_t)N * DIMN;
  float2* Kf  = (float2*)(ws + regA);
  // Region B: Bf / Rf
  float2* Bf = (float2*)(ws + take(szSpec));
  // Region C: Kmat+Vmat (phase 1) then r_ln hi/lo splits (phase 2)
  size_t regC = take(2 * szMat);
  float* Kmat = (float*)(ws + regC);
  float* Vmat = Kmat + (size_t)N * DIMN;
  __bf16* rhi = (__bf16*)(ws + regC);
  __bf16* rlo = rhi + (size_t)N * DIMN;
  // Weight splits
  __bf16* wkhi = (__bf16*)(ws + take((size_t)DIMN * DIMN * 2));
  __bf16* wklo = (__bf16*)(ws + take((size_t)DIMN * DIMN * 2));
  __bf16* wvhi = (__bf16*)(ws + take((size_t)DIMN * DIMN * 2));
  __bf16* wvlo = (__bf16*)(ws + take((size_t)DIMN * DIMN * 2));
  __bf16* wohi = (__bf16*)(ws + take((size_t)DIMN * DIMN * 2));
  __bf16* wolo = (__bf16*)(ws + take((size_t)DIMN * DIMN * 2));
  float2* csum = (float2*)(ws + take((size_t)Bz * NCHNK * NBINS * sizeof(float2)));

  int nx = N * DIMN, nw = DIMN * DIMN;
  k_split_bf16<<<(nx + 255) / 256, 256, 0, stream>>>(x, xhi, xlo, nx);
  k_split_bf16<<<(nw + 255) / 256, 256, 0, stream>>>(Wk, wkhi, wklo, nw);
  k_split_bf16<<<(nw + 255) / 256, 256, 0, stream>>>(Wv, wvhi, wvlo, nw);
  k_split_bf16<<<(nw + 255) / 256, 256, 0, stream>>>(Wo, wohi, wolo, nw);

  dim3 gg(N / 64, DIMN / 64);
  k_gemm_split<<<gg, 256, 0, stream>>>(xhi, xlo, wkhi, wklo, bk, nullptr, Kmat);
  k_gemm_split<<<gg, 256, 0, stream>>>(xhi, xlo, wvhi, wvlo, bv, nullptr, Vmat);

  k_fft_bind<<<N, 256, 0, stream>>>(Kmat, Vmat, Kf, Bf);

  dim3 gs(3, NCHNK, Bz);
  k_scan_chunk_sum<<<gs, 256, 0, stream>>>(Bf, csum, L);
  k_scan_chunk_prefix<<<dim3(3, 1, Bz), 256, 0, stream>>>(csum);
  k_scan_apply_unbind<<<gs, 256, 0, stream>>>(Bf, Kf, csum, L);

  k_ifft_ln<<<N, 256, 0, stream>>>(Bf, lng, lnb, rhi, rlo, L);

  k_gemm_split<<<gg, 256, 0, stream>>>(rhi, rlo, wohi, wolo, bo, x,
                                       (float*)d_out);
}
